// GAT_62663572848804
// MI455X (gfx1250) — matrix operand order repaired and checked by reading, not compiled, not run on hardware
//
#include <hip/hip_runtime.h>

typedef __attribute__((ext_vector_type(16))) __bf16 bf16x16;
typedef __attribute__((ext_vector_type(8)))  __bf16 bf16x8;
typedef __attribute__((ext_vector_type(8)))  float  floatx8;

#define N_NODES 50000
#define IN_DIM  256
#define D1      256   // H1 * HID
#define D2      128   // H2 * NCLS
#define H_HEADS 8
#define C1      32
#define C2      16
#define NEDGE   800000
#define NB      4

// ---------- helpers ----------
__device__ __forceinline__ unsigned short f2bf(float f) {
  unsigned u = __float_as_uint(f);
  u += 0x7FFFu + ((u >> 16) & 1u);      // round-to-nearest-even
  return (unsigned short)(u >> 16);
}

__device__ __forceinline__ void atomicMaxF(float* addr, float val) {
  int* ia = (int*)addr;
  int cur = *ia;
  while (__int_as_float(cur) < val) {
    int prev = atomicCAS(ia, cur, __float_as_int(val));
    if (prev == cur) break;
    cur = prev;
  }
}

__device__ __forceinline__ float lrelu02(float t) { return t > 0.f ? t : 0.2f * t; }

// ---------- cast / transpose ----------
__global__ void cast_bf16_kernel(const float* __restrict__ in,
                                 unsigned short* __restrict__ out, int n) {
  int i = blockIdx.x * blockDim.x + threadIdx.x;
  if (i < n) out[i] = f2bf(in[i]);
}

// W [K, Ncol] row-major  ->  Wt [Ncol, K] row-major, bf16
__global__ void transpose_cast_kernel(const float* __restrict__ W,
                                      unsigned short* __restrict__ Wt,
                                      int K, int Ncol) {
  int i = blockIdx.x * blockDim.x + threadIdx.x;   // over Ncol*K
  if (i >= K * Ncol) return;
  int ncol = i / K, k = i % K;
  Wt[i] = f2bf(W[(size_t)k * Ncol + ncol]);
}

// ---------- WMMA GEMM fragment helpers ----------
struct Frag {
  bf16x8 a0, a1;
  bf16x8 b0[NB], b1[NB];
};

__device__ __forceinline__ void load_frag(Frag& f, const __bf16* __restrict__ arow,
                                          const __bf16* const* brow, int k) {
  f.a0 = *(const bf16x8*)(arow + k);
  f.a1 = *(const bf16x8*)(arow + k + 16);
#pragma unroll
  for (int j = 0; j < NB; ++j) {
    f.b0[j] = *(const bf16x8*)(brow[j] + k);
    f.b1[j] = *(const bf16x8*)(brow[j] + k + 16);
  }
}

__device__ __forceinline__ void mma_frag(floatx8* acc, const Frag& f) {
  bf16x16 av = __builtin_shufflevector(f.a0, f.a1, 0,1,2,3,4,5,6,7,8,9,10,11,12,13,14,15);
#pragma unroll
  for (int j = 0; j < NB; ++j) {
    bf16x16 bv = __builtin_shufflevector(f.b0[j], f.b1[j],
                                         0,1,2,3,4,5,6,7,8,9,10,11,12,13,14,15);
    acc[j] = __builtin_amdgcn_wmma_f32_16x16x32_bf16(false, av, false, bv,
                                                     (short)0, acc[j], false, false);
  }
}

// ---------- WMMA bf16 GEMM:  C[M,Ncol] = A[M,K] * Bt[Ncol,K]^T ----------
// 16x64 strip per wave (1 A-tile x 4 B-tiles). k-loop unrolled x2 with two
// named fragment sets ping-ponging (no inter-set assignments -> no rotation
// copies): loads for step k+1 are issued into the idle set while the 4 WMMAs
// run on the live set. K must be a multiple of 64 (256 here).
__global__ void __launch_bounds__(128)
__attribute__((amdgpu_waves_per_eu(1)))
gemm_bf16_wmma_kernel(const unsigned short* __restrict__ Au,
                      const unsigned short* __restrict__ Bu,
                      float* __restrict__ C, int M, int K, int Ncol) {
  const __bf16* A  = (const __bf16*)Au;
  const __bf16* Bt = (const __bf16*)Bu;
  const int lane    = threadIdx.x & 31;
  const int wave    = threadIdx.x >> 5;
  const int tilesNB = Ncol >> 6;                    // groups of NB=4 N-tiles
  const int tile    = blockIdx.x * 4 + wave;        // wave-uniform guard -> EXEC all-1s
  if (tile >= (M >> 4) * tilesNB) return;
  const int tileM  = tile / tilesNB;
  const int tileNB = tile % tilesNB;
  const int ln = lane & 15;                         // A row / B column within tile
  const int hi = lane >> 4;                         // lane-group select
  // A: group0 K 0..7 & 16..23 ; group1 K 8..15 & 24..31 (ISA 16-bit A 16x32 layout)
  const __bf16* arow = A + (size_t)(tileM * 16 + ln) * K + hi * 8;
  // B: group0 K 0..7 & 16..23 ; group1 K 8..15 & 24..31 (same order as A)
  const __bf16* brow[NB];
#pragma unroll
  for (int j = 0; j < NB; ++j)
    brow[j] = Bt + (size_t)(tileNB * 64 + j * 16 + ln) * K + hi * 8;

  Frag fA, fB;
  load_frag(fA, arow, brow, 0);                     // prologue: k=0 into set A

  floatx8 acc[NB] = {};
  for (int k = 0; k < K; k += 64) {                 // two 32-wide k-steps per iter
    __builtin_prefetch(arow + k + 256, 0, 1);       // global_prefetch_b8
    const int k1 = (k + 32 < K) ? k + 32 : 0;       // wrap keeps addresses valid
    const int k2 = (k + 64 < K) ? k + 64 : 0;
    load_frag(fB, arow, brow, k1);                  // fill idle set B
    mma_frag(acc, fA);                              // 4 WMMAs on live set A
    load_frag(fA, arow, brow, k2);                  // refill set A
    mma_frag(acc, fB);                              // 4 WMMAs on set B
  }
  // C/D f32 16x16 layout: lane<16 -> rows 0..7 (vgpr r = row r), lane>=16 -> rows 8..15
  float* cbase = C + (size_t)(tileM * 16 + hi * 8) * Ncol + tileNB * 64 + ln;
#pragma unroll
  for (int j = 0; j < NB; ++j) {
    float* cp = cbase + j * 16;
#pragma unroll
    for (int r = 0; r < 8; ++r) cp[(size_t)r * Ncol] = acc[j][r];
  }
}

// ---------- attention coefficients + self-loop max init ----------
__global__ void attn_coef_kernel(const float* __restrict__ h,
                                 const float* __restrict__ a_src,
                                 const float* __restrict__ a_dst,
                                 float* __restrict__ as, float* __restrict__ ad,
                                 float* __restrict__ emax, float* __restrict__ den,
                                 int n_nodes, int H, int C) {
  int i = blockIdx.x * blockDim.x + threadIdx.x;   // over n_nodes*H
  if (i >= n_nodes * H) return;
  int n = i / H, hh = i % H;
  const float* hp  = h + (size_t)n * H * C + (size_t)hh * C;
  const float* asp = a_src + hh * C;
  const float* adp = a_dst + hh * C;
  float s = 0.f, d = 0.f;
  for (int c = 0; c < C; ++c) { float v = hp[c]; s += v * asp[c]; d += v * adp[c]; }
  as[i] = s; ad[i] = d;
  emax[i] = lrelu02(s + d);                        // self-loop edge seeds the max
  den[i]  = 0.f;
}

__global__ void edge_max_kernel(const long long* __restrict__ src,
                                const long long* __restrict__ dst,
                                const float* __restrict__ as, const float* __restrict__ ad,
                                float* __restrict__ emax, int nedge, int H) {
  int e = blockIdx.x * blockDim.x + threadIdx.x;
  if (e >= nedge) return;
  int s = (int)src[e], d = (int)dst[e];
  for (int h = 0; h < H; ++h)
    atomicMaxF(&emax[d * H + h], lrelu02(as[s * H + h] + ad[d * H + h]));
}

// covers E real edges + N self loops
__global__ void edge_sum_kernel(const long long* __restrict__ src,
                                const long long* __restrict__ dst,
                                const float* __restrict__ as, const float* __restrict__ ad,
                                const float* __restrict__ emax, float* __restrict__ den,
                                int nedge, int n_nodes, int H) {
  int e = blockIdx.x * blockDim.x + threadIdx.x;
  if (e >= nedge + n_nodes) return;
  int s, d;
  if (e < nedge) { s = (int)src[e]; d = (int)dst[e]; } else { s = d = e - nedge; }
  for (int h = 0; h < H; ++h) {
    float t = lrelu02(as[s * H + h] + ad[d * H + h]);
    atomicAdd(&den[d * H + h], __expf(t - emax[d * H + h]));
  }
}

__global__ void init_bias_kernel(float* __restrict__ out, const float* __restrict__ b,
                                 int n_nodes, int D) {
  int i = blockIdx.x * blockDim.x + threadIdx.x;
  if (i < n_nodes * D) out[i] = b[i % D];
}

// one block per (edge|self-loop), one thread per output channel
__global__ void aggregate_kernel(const long long* __restrict__ src,
                                 const long long* __restrict__ dst,
                                 const float* __restrict__ h,
                                 const float* __restrict__ as, const float* __restrict__ ad,
                                 const float* __restrict__ emax, const float* __restrict__ den,
                                 float* __restrict__ out, int nedge, int H, int C) {
  int e = blockIdx.x;
  int c = threadIdx.x;                              // < H*C
  int s, d;
  if (e < nedge) { s = (int)src[e]; d = (int)dst[e]; } else { s = d = e - nedge; }
  int hh = c / C;
  float t = lrelu02(as[s * H + hh] + ad[d * H + hh]);
  float p = __expf(t - emax[d * H + hh]);
  float alpha = p / (den[d * H + hh] + 1e-16f);
  atomicAdd(&out[(size_t)d * (H * C) + c], alpha * h[(size_t)s * (H * C) + c]);
}

__global__ void elu_cast_kernel(const float* __restrict__ in,
                                unsigned short* __restrict__ outb, int n) {
  int i = blockIdx.x * blockDim.x + threadIdx.x;
  if (i >= n) return;
  float v = in[i];
  outb[i] = f2bf(v > 0.f ? v : __expf(v) - 1.f);
}

// one wave32 per node
__global__ void log_softmax_kernel(const float* __restrict__ in,
                                   float* __restrict__ out, int n_nodes, int D) {
  int n = blockIdx.x * (blockDim.x >> 5) + (threadIdx.x >> 5);
  int lane = threadIdx.x & 31;
  if (n >= n_nodes) return;
  const float* p = in + (size_t)n * D;
  float m = -3.4e38f;
  for (int c = lane; c < D; c += 32) m = fmaxf(m, p[c]);
  for (int o = 16; o > 0; o >>= 1) m = fmaxf(m, __shfl_xor(m, o, 32));
  float s = 0.f;
  for (int c = lane; c < D; c += 32) s += __expf(p[c] - m);
  for (int o = 16; o > 0; o >>= 1) s += __shfl_xor(s, o, 32);
  float lse = __logf(s) + m;
  for (int c = lane; c < D; c += 32) out[(size_t)n * D + c] = p[c] - lse;
}

// ---------- launch ----------
extern "C" void kernel_launch(void* const* d_in, const int* in_sizes, int n_in,
                              void* d_out, int out_size, void* d_ws, size_t ws_size,
                              hipStream_t stream) {
  const float*     x   = (const float*)d_in[0];
  const long long* ei  = (const long long*)d_in[1];   // int64 [2, E]
  const float*     W1  = (const float*)d_in[2];
  const float*     aS1 = (const float*)d_in[3];
  const float*     aD1 = (const float*)d_in[4];
  const float*     b1  = (const float*)d_in[5];
  const float*     W2  = (const float*)d_in[6];
  const float*     aS2 = (const float*)d_in[7];
  const float*     aD2 = (const float*)d_in[8];
  const float*     b2  = (const float*)d_in[9];
  const long long* src = ei;
  const long long* dst = ei + NEDGE;
  float* out = (float*)d_out;

  char* ws = (char*)d_ws;
  size_t off = 0;
  auto alloc = [&](size_t bytes) -> void* {
    void* p = ws + off;
    off += (bytes + 255) & ~(size_t)255;
    return p;
  };
  unsigned short* xb  = (unsigned short*)alloc((size_t)N_NODES * IN_DIM * 2); // later: ELU(h) bf16
  unsigned short* W1t = (unsigned short*)alloc((size_t)IN_DIM * D1 * 2);
  unsigned short* W2t = (unsigned short*)alloc((size_t)D1 * D2 * 2);
  float* h1   = (float*)alloc((size_t)N_NODES * D1 * 4);  // later: [h2 | out2]
  float* out1 = (float*)alloc((size_t)N_NODES * D1 * 4);
  float* as1  = (float*)alloc((size_t)N_NODES * H_HEADS * 4);
  float* ad1  = (float*)alloc((size_t)N_NODES * H_HEADS * 4);
  float* mx1  = (float*)alloc((size_t)N_NODES * H_HEADS * 4);
  float* dn1  = (float*)alloc((size_t)N_NODES * H_HEADS * 4);
  float* as2  = (float*)alloc((size_t)N_NODES * H_HEADS * 4);
  float* ad2  = (float*)alloc((size_t)N_NODES * H_HEADS * 4);
  float* mx2  = (float*)alloc((size_t)N_NODES * H_HEADS * 4);
  float* dn2  = (float*)alloc((size_t)N_NODES * H_HEADS * 4);
  float* h2   = h1;                                   // aliases (h1 dead by then)
  float* out2 = h1 + (size_t)N_NODES * D2;
  unsigned short* hb = xb;                            // alias (xb dead after GEMM1)

  const int B = 256;
  auto cdiv = [](long long a, long long b) { return (int)((a + b - 1) / b); };

  // ===== layer 1 =====
  cast_bf16_kernel<<<cdiv((long long)N_NODES * IN_DIM, B), B, 0, stream>>>(x, xb, N_NODES * IN_DIM);
  transpose_cast_kernel<<<cdiv(IN_DIM * D1, B), B, 0, stream>>>(W1, W1t, IN_DIM, D1);
  transpose_cast_kernel<<<cdiv(D1 * D2, B), B, 0, stream>>>(W2, W2t, D1, D2);

  int tiles1 = (N_NODES / 16) * (D1 / 64);            // 16x64 strips, 4 waves/block
  gemm_bf16_wmma_kernel<<<cdiv(tiles1, 4), 128, 0, stream>>>(xb, W1t, h1, N_NODES, IN_DIM, D1);

  attn_coef_kernel<<<cdiv((long long)N_NODES * H_HEADS, B), B, 0, stream>>>(
      h1, aS1, aD1, as1, ad1, mx1, dn1, N_NODES, H_HEADS, C1);
  edge_max_kernel<<<cdiv(NEDGE, B), B, 0, stream>>>(src, dst, as1, ad1, mx1, NEDGE, H_HEADS);
  edge_sum_kernel<<<cdiv(NEDGE + N_NODES, B), B, 0, stream>>>(
      src, dst, as1, ad1, mx1, dn1, NEDGE, N_NODES, H_HEADS);
  init_bias_kernel<<<cdiv((long long)N_NODES * D1, B), B, 0, stream>>>(out1, b1, N_NODES, D1);
  aggregate_kernel<<<NEDGE + N_NODES, D1, 0, stream>>>(
      src, dst, h1, as1, ad1, mx1, dn1, out1, NEDGE, H_HEADS, C1);
  elu_cast_kernel<<<cdiv((long long)N_NODES * D1, B), B, 0, stream>>>(out1, hb, N_NODES * D1);

  // ===== layer 2 =====
  int tiles2 = (N_NODES / 16) * (D2 / 64);            // 6250 strips
  gemm_bf16_wmma_kernel<<<cdiv(tiles2, 4), 128, 0, stream>>>(hb, W2t, h2, N_NODES, D1, D2);

  attn_coef_kernel<<<cdiv((long long)N_NODES * H_HEADS, B), B, 0, stream>>>(
      h2, aS2, aD2, as2, ad2, mx2, dn2, N_NODES, H_HEADS, C2);
  edge_max_kernel<<<cdiv(NEDGE, B), B, 0, stream>>>(src, dst, as2, ad2, mx2, NEDGE, H_HEADS);
  edge_sum_kernel<<<cdiv(NEDGE + N_NODES, B), B, 0, stream>>>(
      src, dst, as2, ad2, mx2, dn2, NEDGE, N_NODES, H_HEADS);
  init_bias_kernel<<<cdiv((long long)N_NODES * D2, B), B, 0, stream>>>(out2, b2, N_NODES, D2);
  aggregate_kernel<<<NEDGE + N_NODES, D2, 0, stream>>>(
      src, dst, h2, as2, ad2, mx2, dn2, out2, NEDGE, H_HEADS, C2);

  log_softmax_kernel<<<cdiv(N_NODES, 4), 128, 0, stream>>>(out2, out, N_NODES, D2);
}
